// LSTM_custom_56315611186004
// MI455X (gfx1250) — matrix operand order repaired and checked by reading, not compiled, not run on hardware
//
#include <hip/hip_runtime.h>
#include <math.h>

// ---------------------------------------------------------------------------
// LSTM  B=32, T=512, D=1024, H=1024 — bf16 WMMA recurrence, fp32 accumulate.
// Per-WG weight slices cached in CDNA5's 320KB LDS; h broadcast via L2.
// K-loop software-pipelined in registers (double-buffered operand tiles).
// ---------------------------------------------------------------------------

typedef __attribute__((ext_vector_type(8)))  __bf16 bf16x8;
typedef __attribute__((ext_vector_type(16))) __bf16 bf16x16;
typedef __attribute__((ext_vector_type(8)))  float  f32x8;

#define B_  32
#define T_  512
#define D_  1024
#define H_  1024
#define G4_ 4096          // 4*H
#define NWG 64            // persistent workgroups (one 16-col chunk of H each)
#define WPAD 1032         // LDS column stride (elems): 16B-aligned, bank-staggered
#define DYN_LDS_BYTES (2 * 64 * WPAD * 2)   // two matrices x 64 cols x WPAD bf16

// workspace layout (bytes)
static const size_t OFF_XBF  = 0;                                  // 32 MB  (B*T*D bf16)
static const size_t OFF_WIHT = OFF_XBF  + (size_t)B_*T_*D_*2;      // 8 MB   (w_ih^T bf16 [4096][1024])
static const size_t OFF_WHHT = OFF_WIHT + (size_t)G4_*D_*2;        // 8 MB   (w_hh^T bf16 [4096][1024])
static const size_t OFF_BIAS = OFF_WHHT + (size_t)G4_*H_*2;        // 16 KB  (b_ih+b_hh fp32)
static const size_t OFF_HBF  = OFF_BIAS + (size_t)G4_*4;           // 128 KB (2 x [32][1024] bf16, double buffered)
static const size_t OFF_CNT  = OFF_HBF  + (size_t)2*B_*H_*2;       // 2 KB   (per-step barrier counters)

static __device__ __forceinline__ unsigned short f2bf(float f) {
    union { float f; unsigned u; } v; v.f = f;
    unsigned r = v.u + 0x7fffu + ((v.u >> 16) & 1u);   // round-to-nearest-even
    return (unsigned short)(r >> 16);
}

static __device__ __forceinline__ bf16x16 cat8(bf16x8 lo, bf16x8 hi) {
    return __builtin_shufflevector(lo, hi, 0,1,2,3,4,5,6,7,8,9,10,11,12,13,14,15);
}
// A-tile fragment: per-lane chunks at +0 and +16 elements (16B each)
static __device__ __forceinline__ bf16x16 lda(const __bf16* p) {
    return cat8(*(const bf16x8*)p, *(const bf16x8*)(p + 16));
}
// B-tile fragment: per-lane chunks at +0 and +16 elements (16B each)
static __device__ __forceinline__ bf16x16 ldb(const __bf16* p) {
    return cat8(*(const bf16x8*)p, *(const bf16x8*)(p + 16));
}

// ---------------------------------------------------------------------------
// Prep: fp32 -> bf16 conversions, weight transposes, bias sum, h0 seed,
// barrier-counter reset.  Runs every call (harness does not re-zero ws).
// ---------------------------------------------------------------------------
__global__ void lstm_prep(const float* __restrict__ x,
                          const float* __restrict__ h0,
                          const float* __restrict__ w_ih,
                          const float* __restrict__ w_hh,
                          const float* __restrict__ b_ih,
                          const float* __restrict__ b_hh,
                          unsigned short* __restrict__ xbf,
                          unsigned short* __restrict__ wihT,
                          unsigned short* __restrict__ whhT,
                          float* __restrict__ bias,
                          unsigned short* __restrict__ hbf0,
                          unsigned* __restrict__ cnt) {
    const size_t stride = (size_t)gridDim.x * blockDim.x;
    const size_t i0     = (size_t)blockIdx.x * blockDim.x + threadIdx.x;

    for (size_t i = i0; i < (size_t)B_*T_*D_; i += stride)
        xbf[i] = f2bf(x[i]);

    // transpose [1024][4096] -> [4096][1024] (contiguous-K for WMMA B tiles);
    // strided reads are L2-resident (16 MB source), one-time cost.
    for (size_t i = i0; i < (size_t)G4_*D_; i += stride) {
        size_t n = i >> 10, k = i & 1023;
        wihT[i] = f2bf(w_ih[k * G4_ + n]);
        whhT[i] = f2bf(w_hh[k * G4_ + n]);
    }
    for (size_t i = i0; i < (size_t)G4_; i += stride)
        bias[i] = b_ih[i] + b_hh[i];
    for (size_t i = i0; i < (size_t)B_*H_; i += stride)
        hbf0[i] = f2bf(h0[i]);
    for (size_t i = i0; i < (size_t)T_; i += stride)
        cnt[i] = 0u;
}

// ---------------------------------------------------------------------------
// Persistent scan kernel: 64 WGs x 256 threads (8 wave32).
// WG g owns H columns [16g, 16g+16).  Wave w: gate = w>>1, batch-half = w&1.
// Weight slices (64 cols x K=1024, both matrices, 256KB bf16) live in LDS for
// all 512 steps.  Per step each wave runs 2 GEMMs (x_t@w_ih + h@w_hh) into one
// f32 accumulator with v_wmma_f32_16x16x32_bf16; gates combine through LDS;
// cell state persists in LDS; h double-buffered in global bf16; one atomic
// device barrier per step.
// ---------------------------------------------------------------------------
__launch_bounds__(256)
__global__ void lstm_scan(const unsigned short* __restrict__ xbf,
                          const unsigned short* __restrict__ wihT,
                          const unsigned short* __restrict__ whhT,
                          const float* __restrict__ bias,
                          unsigned short* __restrict__ hbf,   // 2 x [32][1024]
                          const float* __restrict__ c0,
                          unsigned* __restrict__ cnt,
                          float* __restrict__ out,            // [32][512][1024]
                          float* __restrict__ hf,             // [32][1024]
                          float* __restrict__ cf) {           // [32][1024]
    const int g    = blockIdx.x;          // 0..63  -> H columns [16g,16g+16)
    const int tid  = threadIdx.x;
    const int wave = tid >> 5;
    const int lane = tid & 31;
    const int mh   = wave & 1;            // batch half (rows 0-15 / 16-31)
    const int gate = wave >> 1;           // 0=i 1=f 2=g 3=o
    const int m0   = mh * 16;
    const int n0   = gate * H_ + g * 16;  // column in the 4H gate space

    __shared__ float ldsg[4][32][17];     // gate preactivations (pad 17: no bank conflicts)
    __shared__ float ldsc[32][17];        // persistent cell state chunk
    extern __shared__ __align__(16) char dynsmem[];
    __bf16* ldsWhh = (__bf16*)dynsmem;            // [64][WPAD]
    __bf16* ldsWih = ldsWhh + 64 * WPAD;          // [64][WPAD]

    // ---- one-time fill: this WG's weight slices into LDS (off critical path)
    for (int e = tid; e < 64 * (H_ / 8); e += 256) {          // 8192 x bf16x8 chunks
        int lc = e >> 7, ch = (e & 127) << 3;                 // local col, K offset
        int n  = (lc >> 4) * H_ + g * 16 + (lc & 15);         // global gate column
        *(bf16x8*)(ldsWhh + lc * WPAD + ch) =
            *(const bf16x8*)((const __bf16*)whhT + (size_t)n * H_ + ch);
        *(bf16x8*)(ldsWih + lc * WPAD + ch) =
            *(const bf16x8*)((const __bf16*)wihT + (size_t)n * D_ + ch);
    }
    for (int e = tid; e < B_ * 16; e += 256) {
        int m = e >> 4, cc = e & 15;
        ldsc[m][cc] = c0[m * H_ + g * 16 + cc];
    }
    __syncthreads();

    // ISA VGPR layouts (CDNA5 ISA):
    //  A 16x32 bf16: lane<16 row=lane  K {0..7,16..23}; lane>=16 row=lane-16 K {8..15,24..31}
    //  B 32x16 bf16: lane<16 col=lane  K {0..7,16..23}; lane>=16 col=lane-16 K {8..15,24..31}
    //  C 16x16 f32 : lane%16 = col; VGPR r -> row r (lane<16) / row 8+r (lane>=16)
    const int arow = lane & 15;
    const int aK   = (lane < 16) ? 0 : 8;
    const int bK   = (lane < 16) ? 0 : 8;

    const __bf16* bHh  = ldsWhh + (gate * 16 + arow) * WPAD + bK;
    const __bf16* bIh  = ldsWih + (gate * 16 + arow) * WPAD + bK;
    const __bf16* xrow = (const __bf16*)xbf + (size_t)(m0 + arow) * T_ * D_ + aK;
    const float   bn   = bias[n0 + arow];

    for (int t = 0; t < T_; ++t) {
        const __bf16* pH = (const __bf16*)hbf + (size_t)(t & 1) * (B_ * H_)
                           + (size_t)(m0 + arow) * H_ + aK;
        const __bf16* pX = xrow + (size_t)t * D_;

        // prefetch next timestep's x row (global_prefetch_b8) to hide HBM/L2
        // latency behind this step's WMMA work
        if (t + 1 < T_) {
            const char* nx = (const char*)(pX + D_);
            #pragma unroll
            for (int p = 0; p < D_ * 2; p += 256)
                __builtin_prefetch(nx + p, 0, 1);
        }

        // ---- K loop, software pipelined: tiles for iteration k+1 are loaded
        // before the WMMAs consuming iteration k, so the compiler can emit
        // partial s_wait counts (overlap loads with matrix math).
        f32x8 acc = {};
        bf16x16 aH = lda(pH), bH = ldb(bHh), aX = lda(pX), bI = ldb(bIh);
        #pragma unroll 2
        for (int kk = 1; kk < H_ / 32; ++kk) {
            const int k = kk * 32;
            bf16x16 aHn = lda(pH + k);
            bf16x16 bHn = ldb(bHh + k);
            bf16x16 aXn = lda(pX + k);
            bf16x16 bIn = ldb(bIh + k);
            acc = __builtin_amdgcn_wmma_f32_16x16x32_bf16(
                      false, aH, false, bH, (short)0, acc, false, false);
            acc = __builtin_amdgcn_wmma_f32_16x16x32_bf16(
                      false, aX, false, bI, (short)0, acc, false, false);
            aH = aHn; bH = bHn; aX = aXn; bI = bIn;
        }
        acc = __builtin_amdgcn_wmma_f32_16x16x32_bf16(
                  false, aH, false, bH, (short)0, acc, false, false);
        acc = __builtin_amdgcn_wmma_f32_16x16x32_bf16(
                  false, aX, false, bI, (short)0, acc, false, false);

        // spill preactivations (+bias) to LDS for cross-wave gate combine
        #pragma unroll
        for (int r = 0; r < 8; ++r) {
            int row = m0 + ((lane >> 4) << 3) + r;
            ldsg[gate][row][arow] = acc[r] + bn;
        }
        __syncthreads();

        // elementwise LSTM cell update: 512 cells / 256 threads (2 per thread)
        unsigned short* hwr = hbf + (size_t)((t + 1) & 1) * (B_ * H_);
        #pragma unroll
        for (int ii = 0; ii < 2; ++ii) {
            int e = tid + ii * 256;
            int m = e >> 4, cc = e & 15;
            float iv = ldsg[0][m][cc], fv = ldsg[1][m][cc];
            float gv = ldsg[2][m][cc], ov = ldsg[3][m][cc];
            float si = 1.f / (1.f + expf(-iv));
            float sf = 1.f / (1.f + expf(-fv));
            float so = 1.f / (1.f + expf(-ov));
            float cn = sf * ldsc[m][cc] + si * tanhf(gv);
            float h  = so * tanhf(cn);
            ldsc[m][cc] = cn;
            int j = g * 16 + cc;
            out[((size_t)m * T_ + t) * H_ + j] = h;
            hwr[(size_t)m * H_ + j] = f2bf(h);
            if (t == T_ - 1) { hf[(size_t)m*H_ + j] = h; cf[(size_t)m*H_ + j] = cn; }
        }

        // device-wide barrier: h(t+1) must be visible before any WG starts t+1
        __threadfence();
        __syncthreads();
        if (tid == 0) {
            __hip_atomic_fetch_add(&cnt[t], 1u, __ATOMIC_RELEASE,
                                   __HIP_MEMORY_SCOPE_AGENT);
            while (__hip_atomic_load(&cnt[t], __ATOMIC_ACQUIRE,
                                     __HIP_MEMORY_SCOPE_AGENT) < (unsigned)NWG)
                __builtin_amdgcn_s_sleep(8);
        }
        __syncthreads();
    }
}

// ---------------------------------------------------------------------------
extern "C" void kernel_launch(void* const* d_in, const int* in_sizes, int n_in,
                              void* d_out, int out_size, void* d_ws, size_t ws_size,
                              hipStream_t stream) {
    const float* x    = (const float*)d_in[0];
    const float* h0   = (const float*)d_in[1];
    const float* c0   = (const float*)d_in[2];
    const float* w_ih = (const float*)d_in[3];
    const float* w_hh = (const float*)d_in[4];
    const float* b_ih = (const float*)d_in[5];
    const float* b_hh = (const float*)d_in[6];

    char* ws = (char*)d_ws;
    unsigned short* xbf  = (unsigned short*)(ws + OFF_XBF);
    unsigned short* wihT = (unsigned short*)(ws + OFF_WIHT);
    unsigned short* whhT = (unsigned short*)(ws + OFF_WHHT);
    float*          bias = (float*)         (ws + OFF_BIAS);
    unsigned short* hbf  = (unsigned short*)(ws + OFF_HBF);
    unsigned*       cnt  = (unsigned*)      (ws + OFF_CNT);

    float* out = (float*)d_out;                      // [32][512][1024]
    float* hf  = out + (size_t)B_ * T_ * H_;         // [32][1024]
    float* cf  = hf + (size_t)B_ * H_;               // [32][1024]

    lstm_prep<<<2048, 256, 0, stream>>>(x, h0, w_ih, w_hh, b_ih, b_hh,
                                        xbf, wihT, whhT, bias, hbf, cnt);
    lstm_scan<<<NWG, 256, DYN_LDS_BYTES, stream>>>(xbf, wihT, whhT, bias, hbf,
                                                   c0, cnt, out, hf, cf);
}
